// ContrastiveLoss_24481313587718
// MI455X (gfx1250) — matrix operand order repaired and checked by reading, not compiled, not run on hardware
//
#include <hip/hip_runtime.h>

#define B_ 128
#define R_ 36
#define L_ 48
#define D_ 1024
#define LAM_S 9.0f
#define LAM_L 6.0f
#define MARGIN_ 0.2f
#define EPS_ 1e-8f

typedef __attribute__((ext_vector_type(16))) __bf16 v16bf;
typedef __attribute__((ext_vector_type(8)))  float  v8f;

static __device__ inline float lrelu(float x) { return x > 0.f ? x : 0.1f * x; }

// A-matrix (16x32 bf16) fragment: lane half selects K groups {0-7,16-23} vs {8-15,24-31}
static __device__ inline v16bf load_a_frag(const __bf16* base, int row0, int nrows, int k) {
  int t = threadIdx.x & 31;
  int sub = t >> 4;
  int m = t & 15;
  int r = row0 + m; if (r >= nrows) r = nrows - 1;
  const __bf16* p = base + (size_t)r * D_ + k + sub * 8;
  v16bf f;
#pragma unroll
  for (int j = 0; j < 8; ++j) f[j] = p[j];
#pragma unroll
  for (int j = 0; j < 8; ++j) f[8 + j] = p[16 + j];
  return f;
}

// B-matrix (32x16 bf16) fragment: lane = N; lane half selects K groups {0-7,16-23} vs {8-15,24-31}
static __device__ inline v16bf load_b_frag(const __bf16* base, int col0, int ncols, int k) {
  int t = threadIdx.x & 31;
  int sub = t >> 4;
  int n = t & 15;
  int c = col0 + n; if (c >= ncols) c = ncols - 1;
  const __bf16* p = base + (size_t)c * D_ + k + sub * 8;
  v16bf f;
#pragma unroll
  for (int j = 0; j < 8; ++j) f[j] = p[j];
#pragma unroll
  for (int j = 0; j < 8; ++j) f[8 + j] = p[16 + j];
  return f;
}

static __device__ inline v8f wmma_bf16(v16bf a, v16bf b, v8f c) {
  // (neg_a, A, neg_b, B, c_mod, C, reuse_a, reuse_b)
  return __builtin_amdgcn_wmma_f32_16x16x32_bf16(false, a, false, b, (short)0, c, false, false);
}

// ---------------------------------------------------------------------------
// Kernel 0: one-time f32 -> bf16 conversion of both inputs into workspace.
// ---------------------------------------------------------------------------
__global__ void cvt_kernel(const float* __restrict__ a, __bf16* __restrict__ oa, int na,
                           const float* __restrict__ b, __bf16* __restrict__ ob, int nb) {
  int idx = blockIdx.x * blockDim.x + threadIdx.x;
  int stride = gridDim.x * blockDim.x;
  for (int t = idx; t < na; t += stride) oa[t] = (__bf16)a[t];
  for (int t = idx; t < nb; t += stride) ob[t] = (__bf16)b[t];
}

// ---------------------------------------------------------------------------
// Kernel 1: Gram matrices (img_gram[i] = images[i] @ images[i]^T, cap_gram likewise)
// One wave per matrix; 3x3 grid of 16x16 WMMA tiles, K=1024.
// ---------------------------------------------------------------------------
__global__ void __launch_bounds__(32, 1)
gram_kernel(const __bf16* __restrict__ images, const __bf16* __restrict__ captions,
            float* __restrict__ img_gram, float* __restrict__ cap_gram) {
  int b = blockIdx.x;
  const __bf16* X; float* G; int nr;
  if (b < B_) { X = images + (size_t)b * R_ * D_;          G = img_gram + (size_t)b * R_ * R_; nr = R_; }
  else        { X = captions + (size_t)(b - B_) * L_ * D_; G = cap_gram + (size_t)(b - B_) * L_ * L_; nr = L_; }

  v8f zero = {0.f,0.f,0.f,0.f,0.f,0.f,0.f,0.f};
  v8f acc[3][3];
#pragma unroll
  for (int a = 0; a < 3; ++a)
#pragma unroll
    for (int bb = 0; bb < 3; ++bb) acc[a][bb] = zero;

  for (int k = 0; k < D_; k += 32) {
    v16bf A0 = load_a_frag(X,  0, nr, k);
    v16bf A1 = load_a_frag(X, 16, nr, k);
    v16bf A2 = load_a_frag(X, 32, nr, k);
    v16bf Bv0 = load_b_frag(X,  0, nr, k);
    v16bf Bv1 = load_b_frag(X, 16, nr, k);
    v16bf Bv2 = load_b_frag(X, 32, nr, k);
    acc[0][0] = wmma_bf16(A0, Bv0, acc[0][0]);
    acc[0][1] = wmma_bf16(A0, Bv1, acc[0][1]);
    acc[0][2] = wmma_bf16(A0, Bv2, acc[0][2]);
    acc[1][0] = wmma_bf16(A1, Bv0, acc[1][0]);
    acc[1][1] = wmma_bf16(A1, Bv1, acc[1][1]);
    acc[1][2] = wmma_bf16(A1, Bv2, acc[1][2]);
    acc[2][0] = wmma_bf16(A2, Bv0, acc[2][0]);
    acc[2][1] = wmma_bf16(A2, Bv1, acc[2][1]);
    acc[2][2] = wmma_bf16(A2, Bv2, acc[2][2]);
  }

  int t = threadIdx.x & 31; int sub = t >> 4; int n = t & 15;
#pragma unroll
  for (int ti = 0; ti < 3; ++ti)
#pragma unroll
    for (int tj = 0; tj < 3; ++tj)
#pragma unroll
      for (int v = 0; v < 8; ++v) {
        int row = ti * 16 + v + sub * 8;
        int col = tj * 16 + n;
        if (row < nr && col < nr) G[row * nr + col] = acc[ti][tj][v];
      }
}

// ---------------------------------------------------------------------------
// Kernel 2: fused per-(image i, caption c) scoring.
// Phase A: dots[48x36] via WMMA (bf16 in, f32 acc) -> LDS.
// Phase B: full t2i + i2t attention pipeline in-wave -> scores[i*128+c].
// ---------------------------------------------------------------------------
__global__ void __launch_bounds__(32, 1)
score_kernel(const __bf16* __restrict__ images, const __bf16* __restrict__ captions,
             const int* __restrict__ cap_lens,
             const float* __restrict__ img_gram, const float* __restrict__ cap_gram,
             float* __restrict__ scores) {
  __shared__ float sd[L_ * 48];   // dots, stride 48 (cols 36..47 unused)
  __shared__ float wk[L_ * 48];   // attention work buffer
  __shared__ float cn[R_];        // column norms (t2i l2norm over words)
  __shared__ float rn[L_];        // row norms (i2t l2norm over regions)
  __shared__ float rowexp[L_];
  __shared__ float colexp[R_];

  const int i = blockIdx.x;   // image
  const int c = blockIdx.y;   // caption
  const int tid = threadIdx.x;
  const int len = cap_lens[c];
  const float nwf = (float)len;

  const __bf16* capb = captions + (size_t)c * L_ * D_;
  const __bf16* imgb = images + (size_t)i * R_ * D_;
  const float* Gi = img_gram + (size_t)i * R_ * R_;
  const float* Gc = cap_gram + (size_t)c * L_ * L_;

  // ---- Phase A: dots = caps[c] @ images[i]^T via WMMA ----
  v8f zero = {0.f,0.f,0.f,0.f,0.f,0.f,0.f,0.f};
  v8f acc[3][3];
#pragma unroll
  for (int a = 0; a < 3; ++a)
#pragma unroll
    for (int bb = 0; bb < 3; ++bb) acc[a][bb] = zero;

  for (int k = 0; k < D_; k += 32) {
    v16bf A0 = load_a_frag(capb,  0, L_, k);
    v16bf A1 = load_a_frag(capb, 16, L_, k);
    v16bf A2 = load_a_frag(capb, 32, L_, k);
    v16bf Bv0 = load_b_frag(imgb,  0, R_, k);
    v16bf Bv1 = load_b_frag(imgb, 16, R_, k);
    v16bf Bv2 = load_b_frag(imgb, 32, R_, k);
    acc[0][0] = wmma_bf16(A0, Bv0, acc[0][0]);
    acc[0][1] = wmma_bf16(A0, Bv1, acc[0][1]);
    acc[0][2] = wmma_bf16(A0, Bv2, acc[0][2]);
    acc[1][0] = wmma_bf16(A1, Bv0, acc[1][0]);
    acc[1][1] = wmma_bf16(A1, Bv1, acc[1][1]);
    acc[1][2] = wmma_bf16(A1, Bv2, acc[1][2]);
    acc[2][0] = wmma_bf16(A2, Bv0, acc[2][0]);
    acc[2][1] = wmma_bf16(A2, Bv1, acc[2][1]);
    acc[2][2] = wmma_bf16(A2, Bv2, acc[2][2]);
  }
  {
    int t = tid & 31; int sub = t >> 4; int n = t & 15;
#pragma unroll
    for (int ti = 0; ti < 3; ++ti)
#pragma unroll
      for (int tj = 0; tj < 3; ++tj)
#pragma unroll
        for (int v = 0; v < 8; ++v)
          sd[(ti * 16 + v + sub * 8) * 48 + tj * 16 + n] = acc[ti][tj][v];
  }
  __syncthreads();

  // ---- t2i: column l2norm over words ----
  for (int r = tid; r < R_; r += 32) {
    float s = 0.f;
    for (int l = 0; l < L_; ++l) { float x = lrelu(sd[l * 48 + r]); s += x * x; }
    cn[r] = sqrtf(s) + EPS_;
  }
  __syncthreads();

  // ---- t2i: per-word softmax over regions + focal + cosine ----
  for (int l = tid; l < L_; l += 32) {
    float mx = -1e30f;
    for (int r = 0; r < R_; ++r) {
      float x = LAM_S * lrelu(sd[l * 48 + r]) / cn[r];
      if (x > mx) mx = x;
    }
    float esum = 0.f;
    for (int r = 0; r < R_; ++r) {
      float x = LAM_S * lrelu(sd[l * 48 + r]) / cn[r];
      float e = __expf(x - mx);
      wk[l * 48 + r] = e; esum += e;
    }
    float satt = 0.f;
    for (int r = 0; r < R_; ++r) satt += wk[l * 48 + r] / esum;
    float ts = 0.f;
    for (int r = 0; r < R_; ++r) {
      float at = wk[l * 48 + r] / esum;
      float keep = (at * (float)R_ - satt > 0.f) ? at : 0.f;
      wk[l * 48 + r] = keep; ts += keep;
    }
    float inv = (ts > 0.f) ? 1.f / ts : 1.f;
    float w12 = 0.f;
    for (int r = 0; r < R_; ++r) {
      float re = wk[l * 48 + r] * inv;
      wk[l * 48 + r] = re;
      w12 += re * sd[l * 48 + r];
    }
    float q = 0.f;
    for (int r = 0; r < R_; ++r) {
      float vr = 0.f;
      for (int s = 0; s < R_; ++s) vr += wk[l * 48 + s] * Gi[r * R_ + s];
      q += wk[l * 48 + r] * vr;
    }
    float w2 = sqrtf(fmaxf(q, 1e-16f));
    float capn = sqrtf(fmaxf(Gc[l * L_ + l], 1e-16f));
    float sim = w12 / fmaxf(capn * w2, EPS_);
    rowexp[l] = (l < len) ? __expf(LAM_L * sim) : 0.f;
  }
  __syncthreads();

  // ---- i2t: row l2norm over regions ----
  for (int l = tid; l < L_; l += 32) {
    float s = 0.f;
    for (int r = 0; r < R_; ++r) { float x = lrelu(sd[l * 48 + r]); s += x * x; }
    rn[l] = sqrtf(s) + EPS_;
  }
  __syncthreads();

  // ---- i2t: per-region softmax over valid words + focal + cosine ----
  for (int r = tid; r < R_; r += 32) {
    float mx = -1e30f;
    for (int l = 0; l < len; ++l) {
      float x = LAM_S * lrelu(sd[l * 48 + r]) / rn[l];
      if (x > mx) mx = x;
    }
    float esum = 0.f;
    for (int l = 0; l < len; ++l) {
      float x = LAM_S * lrelu(sd[l * 48 + r]) / rn[l];
      float e = __expf(x - mx);
      wk[l * 48 + r] = e; esum += e;
    }
    for (int l = len; l < L_; ++l) wk[l * 48 + r] = 0.f;
    float satt = 0.f;
    for (int l = 0; l < len; ++l) satt += wk[l * 48 + r] / esum;
    float ts = 0.f;
    for (int l = 0; l < len; ++l) {
      float at = wk[l * 48 + r] / esum;
      float keep = (at * nwf - satt > 0.f) ? at : 0.f;
      wk[l * 48 + r] = keep; ts += keep;
    }
    float inv = (ts > 0.f) ? 1.f / ts : 1.f;
    float w12 = 0.f;
    for (int l = 0; l < len; ++l) {
      float re = wk[l * 48 + r] * inv;
      wk[l * 48 + r] = re;
      w12 += re * sd[l * 48 + r];
    }
    float q = 0.f;
    for (int l = 0; l < len; ++l) {
      float vl = 0.f;
      for (int m = 0; m < len; ++m) vl += wk[m * 48 + r] * Gc[l * L_ + m];
      q += wk[l * 48 + r] * vl;
    }
    float imgn = sqrtf(fmaxf(Gi[r * R_ + r], 1e-16f));
    float sim = w12 / fmaxf(imgn * sqrtf(fmaxf(q, 1e-16f)), EPS_);
    colexp[r] = __expf(LAM_L * sim);
  }
  __syncthreads();

  if (tid == 0) {
    float st = 0.f, si = 0.f;
    for (int l = 0; l < L_; ++l) st += rowexp[l];
    for (int r = 0; r < R_; ++r) si += colexp[r];
    float t2i = logf(st) / LAM_L;
    float i2t = logf(si) / LAM_L;
    scores[i * B_ + c] = t2i + i2t;   // scores[n_image][n_caption]
  }
}

// ---------------------------------------------------------------------------
// Kernel 3: max-violation margin ranking loss over the 128x128 score matrix.
// ---------------------------------------------------------------------------
__global__ void loss_kernel(const float* __restrict__ scores, float* __restrict__ out) {
  __shared__ float red[B_];
  int j = threadIdx.x;
  float dj = scores[j * B_ + j];
  float rmax = 0.f, cmax = 0.f;
  for (int k = 0; k < B_; ++k) {
    if (k == j) continue;
    float a = MARGIN_ + scores[j * B_ + k] - dj;          // cost_s row j
    if (a < 0.f) a = 0.f; if (a > rmax) rmax = a;
    float b = MARGIN_ + scores[k * B_ + j] - dj;          // cost_im col j
    if (b < 0.f) b = 0.f; if (b > cmax) cmax = b;
  }
  red[j] = rmax + cmax;
  __syncthreads();
  for (int s = B_ / 2; s > 0; s >>= 1) {
    if (j < s) red[j] += red[j + s];
    __syncthreads();
  }
  if (j == 0) out[0] = red[0];
}

extern "C" void kernel_launch(void* const* d_in, const int* in_sizes, int n_in,
                              void* d_out, int out_size, void* d_ws, size_t ws_size,
                              hipStream_t stream) {
  (void)in_sizes; (void)n_in; (void)out_size; (void)ws_size;
  const float* images_f   = (const float*)d_in[0];
  const float* captions_f = (const float*)d_in[1];
  const int*   cap_lens   = (const int*)d_in[2];

  const int n_img = B_ * R_ * D_;   // 4,718,592
  const int n_cap = B_ * L_ * D_;   // 6,291,456

  char* wsb = (char*)d_ws;
  __bf16* img_bf = (__bf16*)wsb;                                   // 9,437,184 B
  __bf16* cap_bf = (__bf16*)(wsb + (size_t)n_img * 2);             // 12,582,912 B
  float*  img_gram = (float*)(wsb + (size_t)(n_img + n_cap) * 2);  // 128*36*36
  float*  cap_gram = img_gram + (size_t)B_ * R_ * R_;              // 128*48*48
  float*  scores   = cap_gram + (size_t)B_ * L_ * L_;              // 128*128

  cvt_kernel<<<2048, 256, 0, stream>>>(images_f, img_bf, n_img, captions_f, cap_bf, n_cap);
  gram_kernel<<<2 * B_, 32, 0, stream>>>(img_bf, cap_bf, img_gram, cap_gram);
  dim3 grid(B_, B_);
  score_kernel<<<grid, 32, 0, stream>>>(img_bf, cap_bf, cap_lens, img_gram, cap_gram, scores);
  loss_kernel<<<1, B_, 0, stream>>>(scores, (float*)d_out);
}
